// GatingAttentionOpti_77575699300524
// MI455X (gfx1250) — matrix operand order repaired and checked by reading, not compiled, not run on hardware
//
#include <hip/hip_runtime.h>

// ---------------------------------------------------------------------------
// Gating attention for MI455X (gfx1250, wave32, WMMA).
// B=128, S=512, D=256, H=8, HD=32, NH=H*HD=256.
// ---------------------------------------------------------------------------

#define BB   128
#define SS   512
#define DD   256
#define HH   8
#define HDIM 32
#define NHH  256   // H*HD

typedef __bf16 bf16;
typedef __attribute__((ext_vector_type(16))) __bf16 v16bf;
typedef __attribute__((ext_vector_type(8)))  __bf16 v8bf;
typedef __attribute__((ext_vector_type(8)))  float  v8f;
typedef __attribute__((ext_vector_type(4)))  float  v4f;

// Load a 16-element bf16 A/B fragment from two contiguous 8-element runs.
__device__ __forceinline__ v16bf ld2x8bf(const bf16* p0, const bf16* p1) {
  v8bf a = *(const v8bf*)p0;
  v8bf b = *(const v8bf*)p1;
  v16bf r;
#pragma unroll
  for (int i = 0; i < 8; ++i) { r[i] = a[i]; r[i + 8] = b[i]; }
  return r;
}

// Load 16 contiguous bf16 (32B) — B-fragment (n = lane&15, k = khalf + i).
__device__ __forceinline__ v16bf ld16bf(const bf16* p) {
  return *(const v16bf*)p;
}

// fp32 -> bf16 fragment from two contiguous 8-float runs.
__device__ __forceinline__ v16bf cvt2x8f(const float* p0, const float* p1) {
  v4f a0 = ((const v4f*)p0)[0], a1 = ((const v4f*)p0)[1];
  v4f b0 = ((const v4f*)p1)[0], b1 = ((const v4f*)p1)[1];
  v16bf r;
#pragma unroll
  for (int i = 0; i < 4; ++i) {
    r[i]      = (bf16)a0[i];
    r[i + 4]  = (bf16)a1[i];
    r[i + 8]  = (bf16)b0[i];
    r[i + 12] = (bf16)b1[i];
  }
  return r;
}

// ---------------------------------------------------------------------------
// Projection GEMM:  Out[M=65536, N=256] = A[M, K=256] * W[K=256, N=256]
// MODE 0: Q   -> bf16 [B,H,S,HD], scaled by HD^-0.5
// MODE 1: K   -> bf16 [B,H,S,HD]
// MODE 2: V   -> bf16 [B,H,HD,S]   (transposed so PV B-frags are contiguous)
// MODE 3: gate-> f32  [B,S,NH]  sigmoid(acc + gating_b)
// MODE 4: out -> f32  [B,S,D]   acc + output_b
// Block: 256 thr (8 waves). 64 output rows/block; wave (mst,nhalf) does
// 16 rows x 128 cols.  Weight K-slice staged fragment-ordered in LDS.
// ---------------------------------------------------------------------------
template <int MODE, bool ABF16>
__global__ __launch_bounds__(256) void proj_gemm(
    const void* __restrict__ Ap, const float* __restrict__ W,
    const float* __restrict__ bvec, void* __restrict__ Out, float scale) {
  __shared__ __align__(32) bf16 ldsb[16 * 32 * 16];  // 16 ntiles x 32 lanes x 16 elems

  const int tid  = threadIdx.x;
  const int lane = tid & 31;
  const int wave = tid >> 5;
  const int mst  = wave & 3;         // 16-row strip
  const int ntb  = (wave >> 2) * 8;  // 8 n-tiles per wave
  const int nloc = lane & 15;
  const int mhi  = (lane >> 4) << 3;
  const int row0 = blockIdx.x * 64;

  v8f acc[8] = {};

  const int am  = row0 + mst * 16 + nloc;  // A-frag row for this lane
  const int kb8 = (lane >> 4) * 8;         // A-frag k base (0 or 8)

  for (int ks = 0; ks < 256; ks += 32) {
    __syncthreads();
    // Stage 32x256 weight slice into LDS, fragment-ordered, bf16.
#pragma unroll 4
    for (int j = 0; j < 32; ++j) {
      int idx = j * 256 + tid;        // coalesced over n
      int k = idx >> 8, n = idx & 255;
      float wv = W[(ks + k) * 256 + n];
      int bl = (((k >> 3) & 1) << 4) | (n & 15);              // B-frag lane
      ldsb[(((n >> 4) * 32 + bl) << 4) + ((k & 7) | ((k >> 4) << 3))] = (bf16)wv;
    }
    __syncthreads();

    v16bf af;
    if constexpr (ABF16) {
      const bf16* A = (const bf16*)Ap;
      const bf16* p = A + (size_t)am * 256 + ks + kb8;
      af = ld2x8bf(p, p + 16);
    } else {
      const float* A = (const float*)Ap;
      const float* p = A + (size_t)am * 256 + ks + kb8;
      af = cvt2x8f(p, p + 16);
    }

#pragma unroll
    for (int t = 0; t < 8; ++t) {
      v16bf bfr = *(const v16bf*)&ldsb[((ntb + t) * 32 + lane) << 4];
      acc[t] = __builtin_amdgcn_wmma_f32_16x16x32_bf16(
          false, af, false, bfr, (short)0, acc[t], false, false);
    }
  }

  // Epilogue: C layout -> lane holds column n, rows j+mhi.
#pragma unroll
  for (int t = 0; t < 8; ++t) {
    const int n = ((ntb + t) << 4) + nloc;
#pragma unroll
    for (int j = 0; j < 8; ++j) {
      const int mg = row0 + mst * 16 + mhi + j;
      float v = acc[t][j];
      if constexpr (MODE == 0 || MODE == 1) {
        if constexpr (MODE == 0) v *= scale;
        const int b_ = mg >> 9, s_ = mg & 511;
        const int h_ = n >> 5, c_ = n & 31;
        ((bf16*)Out)[(((b_ * HH + h_) * SS + s_) << 5) + c_] = (bf16)v;
      } else if constexpr (MODE == 2) {
        const int b_ = mg >> 9, s_ = mg & 511;
        const int h_ = n >> 5, c_ = n & 31;
        ((bf16*)Out)[(((b_ * HH + h_) * HDIM + c_) << 9) + s_] = (bf16)v;
      } else if constexpr (MODE == 3) {
        float g = v + bvec[n];
        ((float*)Out)[((size_t)mg << 8) + n] =
            __builtin_amdgcn_rcpf(1.0f + __expf(-g));
      } else {
        ((float*)Out)[((size_t)mg << 8) + n] = v + bvec[n];
      }
    }
  }
}

// ---------------------------------------------------------------------------
// Flash attention + gating.  One wave owns 16 query rows; streams keys in
// blocks of 32 with online softmax.  Raw logits are transposed (C->A layout)
// through LDS so row max/sum are in-lane reductions + one lane-16 butterfly;
// exp feeds the bf16 P A-fragment directly.  Per-row alpha/рsum go back to
// C layout via a 16-float LDS broadcast.
// qp/kp: bf16 [B,H,S,HD]; vpT: bf16 [B,H,HD,S]; gate: f32 [B,S,NH];
// output gated: bf16 [B*S, NH].  Grid (S/128, H, B), 256 threads.
// ---------------------------------------------------------------------------
#define SROW 36  // padded fp32 row stride (16B-aligned, de-conflicted banks)

__global__ __launch_bounds__(256) void attn_flash(
    const bf16* __restrict__ qp, const bf16* __restrict__ kp,
    const bf16* __restrict__ vpT, const float* __restrict__ bias,
    const float* __restrict__ nbias, const float* __restrict__ gate,
    bf16* __restrict__ gated) {
  __shared__ __align__(16) float sbuf[8][16 * SROW];  // logits transpose buf
  __shared__ float abuf[8][16];                       // alpha / rsum bcast

  const int lane = threadIdx.x & 31;
  const int wave = threadIdx.x >> 5;
  const int nloc = lane & 15;
  const int mhi  = (lane >> 4) << 3;   // 0 or 8
  const int kb8  = (lane >> 4) * 8;    // A-frag k base
  const int ch   = (lane >> 4) * 16;   // B-frag k half
  const int b    = blockIdx.z;
  const int h    = blockIdx.y;
  const int qtile = blockIdx.x * 128 + wave * 16;

  const size_t headQK = (size_t)(b * HH + h) * SS;   // rows of [S,HD]
  const size_t headV  = (size_t)(b * HH + h) * HDIM; // rows of [HD,S]

  // Q A-fragment: m = lane&15, k in {ch..ch+15}.
  const bf16* qrow = qp + ((headQK + qtile + nloc) << 5) + ch;
  v16bf qf = ld2x8bf(qrow, qrow + 8);

  v8f o0 = {}, o1 = {};
  float rmax = -3.0e38f, rsum = 0.0f;  // A layout: row = lane&15

  float* sb = sbuf[wave];
  float* ab = abuf[wave];

  for (int k0 = 0; k0 < SS; k0 += 32) {
    // K^T B-fragments: lane=key, 16 contiguous c (32B loads).
    const bf16* kr = kp + ((headQK + k0 + nloc) << 5) + ch;
    v16bf kf0 = ld16bf(kr);
    v16bf kf1 = ld16bf(kr + (16 << 5));  // keys +16

    v8f z = {};
    v8f s0 = __builtin_amdgcn_wmma_f32_16x16x32_bf16(false, qf, false, kf0,
                                                     (short)0, z, false, false);
    v8f s1 = __builtin_amdgcn_wmma_f32_16x16x32_bf16(false, qf, false, kf1,
                                                     (short)0, z, false, false);

    // bias[b,q,k] + nonbatched_bias[q,k]  (rows coalesced across lanes 0-15)
#pragma unroll
    for (int j = 0; j < 8; ++j) {
      const int q = qtile + mhi + j;
      const float* bp = bias + ((size_t)(b * SS + q) << 9) + k0 + nloc;
      const float* np = nbias + (q << 9) + k0 + nloc;
      s0[j] += bp[0] + np[0];
      s1[j] += bp[16] + np[16];
    }
    if (k0 + 32 < SS)
      __builtin_prefetch(
          bias + ((size_t)(b * SS + qtile + mhi) << 9) + k0 + 32 + lane, 0, 3);

    // Transpose raw logits C->A through LDS (in-wave DS ordering only).
#pragma unroll
    for (int j = 0; j < 8; ++j) {
      const int m = mhi + j;
      sb[m * SROW + nloc]      = s0[j];
      sb[m * SROW + 16 + nloc] = s1[j];
    }
    asm volatile("s_wait_dscnt 0x0" ::: "memory");
    // A layout: row = lane&15; k chunk {ch..ch+15}.
    const float* srow = &sb[nloc * SROW];
    v4f r0 = *(const v4f*)&srow[ch];
    v4f r1 = *(const v4f*)&srow[ch + 4];
    v4f r2 = *(const v4f*)&srow[ch + 8];
    v4f r3 = *(const v4f*)&srow[ch + 12];

    // Row max: in-lane over 16, then combine the two k-halves (lane ^ 16).
    float m16 = r0[0];
#pragma unroll
    for (int i = 0; i < 4; ++i) {
      m16 = fmaxf(m16, r0[i]);
      m16 = fmaxf(m16, r1[i]);
      m16 = fmaxf(m16, r2[i]);
      m16 = fmaxf(m16, r3[i]);
    }
    m16 = fmaxf(m16, __shfl_xor(m16, 16, 32));
    const float nm    = fmaxf(rmax, m16);
    const float alpha = __expf(rmax - nm);
    rmax = nm;

    // P = exp(S - max) directly into the A-fragment; accumulate row sum.
    v16bf pf;
    float ps = 0.0f;
#pragma unroll
    for (int i = 0; i < 4; ++i) {
      float p0 = __expf(r0[i] - nm);
      float p1 = __expf(r1[i] - nm);
      float p2 = __expf(r2[i] - nm);
      float p3 = __expf(r3[i] - nm);
      ps += p0 + p1 + p2 + p3;
      pf[i]      = (bf16)p0;
      pf[i + 4]  = (bf16)p1;
      pf[i + 8]  = (bf16)p2;
      pf[i + 12] = (bf16)p3;
    }
    ps += __shfl_xor(ps, 16, 32);
    rsum = rsum * alpha + ps;

    // Broadcast alpha back to C layout (lanes l and l^16 hold equal values).
    ab[nloc] = alpha;
    asm volatile("s_wait_dscnt 0x0" ::: "memory");
#pragma unroll
    for (int j = 0; j < 8; ++j) {
      const float aC = ab[mhi + j];
      o0[j] *= aC;
      o1[j] *= aC;
    }

    // V B-fragments from transposed layout: lane=c, 16 contiguous keys (32B).
    const bf16* vr = vpT + ((headV + nloc) << 9) + k0 + ch;
    v16bf vf0 = ld16bf(vr);
    v16bf vf1 = ld16bf(vr + (16 << 9));  // c + 16

    o0 = __builtin_amdgcn_wmma_f32_16x16x32_bf16(false, pf, false, vf0,
                                                 (short)0, o0, false, false);
    o1 = __builtin_amdgcn_wmma_f32_16x16x32_bf16(false, pf, false, vf1,
                                                 (short)0, o1, false, false);
  }

  // Broadcast row sums to C layout; normalize, gate, store bf16.
  ab[nloc] = rsum;
  asm volatile("s_wait_dscnt 0x0" ::: "memory");
#pragma unroll
  for (int j = 0; j < 8; ++j) {
    const float inv = __builtin_amdgcn_rcpf(ab[mhi + j]);
    const int q = qtile + mhi + j;
    const size_t gi = ((size_t)(b * SS + q) << 8) + (h << 5);
    const float g0 = gate[gi + nloc];
    const float g1 = gate[gi + 16 + nloc];
    gated[gi + nloc]      = (bf16)(o0[j] * inv * g0);
    gated[gi + 16 + nloc] = (bf16)(o1[j] * inv * g1);
  }
}

// ---------------------------------------------------------------------------
// Launch
// ---------------------------------------------------------------------------
extern "C" void kernel_launch(void* const* d_in, const int* in_sizes, int n_in,
                              void* d_out, int out_size, void* d_ws,
                              size_t ws_size, hipStream_t stream) {
  const float* q_data   = (const float*)d_in[0];
  const float* m_data   = (const float*)d_in[1];
  const float* bias     = (const float*)d_in[2];
  const float* nbias    = (const float*)d_in[3];
  const float* query_w  = (const float*)d_in[4];
  const float* key_w    = (const float*)d_in[5];
  const float* value_w  = (const float*)d_in[6];
  const float* gating_w = (const float*)d_in[7];
  const float* gating_b = (const float*)d_in[8];
  const float* output_w = (const float*)d_in[9];
  const float* output_b = (const float*)d_in[10];

  // Workspace carve-up (bytes):
  //   qp   bf16 [B,H,S,HD]   33,554,432
  //   kp   bf16 [B,H,S,HD]   33,554,432
  //   vpT  bf16 [B,H,HD,S]   33,554,432
  //   gate f32  [B,S,NH]     67,108,864
  //   gated bf16 [B*S,NH]    33,554,432   (total ~192 MiB)
  char* w = (char*)d_ws;
  bf16*  qp    = (bf16*)w;               w += (size_t)33554432;
  bf16*  kp    = (bf16*)w;               w += (size_t)33554432;
  bf16*  vpT   = (bf16*)w;               w += (size_t)33554432;
  float* gateb = (float*)w;              w += (size_t)67108864;
  bf16*  gated = (bf16*)w;

  const float scale = 0.17677669529663687f;  // HD^-0.5

  dim3 blk(256);
  proj_gemm<0, false><<<1024, blk, 0, stream>>>(q_data, query_w,  nullptr,  qp,    scale);
  proj_gemm<1, false><<<1024, blk, 0, stream>>>(m_data, key_w,    nullptr,  kp,    1.0f);
  proj_gemm<2, false><<<1024, blk, 0, stream>>>(m_data, value_w,  nullptr,  vpT,   1.0f);
  proj_gemm<3, false><<<1024, blk, 0, stream>>>(q_data, gating_w, gating_b, gateb, 1.0f);

  attn_flash<<<dim3(4, HH, BB), blk, 0, stream>>>(qp, kp, vpT, bias, nbias,
                                                  gateb, gated);

  proj_gemm<4, true><<<1024, blk, 0, stream>>>(gated, output_w, output_b,
                                               (float*)d_out, 1.0f);
}
